// Masked_Self_Attention_ALiBi_90091234001522
// MI455X (gfx1250) — matrix operand order repaired and checked by reading, not compiled, not run on hardware
//
#include <hip/hip_runtime.h>
#include <hip/hip_bf16.h>

// ---------------------------------------------------------------------------
// Masked self-attention with ALiBi, bf16 WMMA + async-LDS pipeline (gfx1250).
// B=4, S=2048, H=1024, NH=16, D=64
// Structure: (0) f32->bf16 streaming pre-pass for x and all weights,
//            (1) QKV GEMM (pure async-fed WMMA loop),
//            (2) flash attention (async-fed, online softmax),
//            (3) output projection GEMM (pure async-fed WMMA loop).
// ---------------------------------------------------------------------------

typedef __attribute__((ext_vector_type(16))) __bf16 v16bf;
typedef __attribute__((ext_vector_type(2)))  __bf16 v2bf;
typedef __attribute__((ext_vector_type(8)))  float  v8f;

union FragBF { v16bf v; unsigned u[8]; };

#if defined(__has_builtin)
#if __has_builtin(__builtin_amdgcn_cvt_pk_bf16_f32)
#define HAVE_CVT_PK_BF16 1
#endif
#endif

__device__ __forceinline__ unsigned short f2bf(float f) {
    union { float f; unsigned u; } x; x.f = f;
    unsigned r = x.u + 0x7FFFu + ((x.u >> 16) & 1u);   // round-to-nearest-even
    return (unsigned short)(r >> 16);
}

__device__ __forceinline__ unsigned pack_bf16(float a, float b) {
#ifdef HAVE_CVT_PK_BF16
    union { v2bf v; unsigned u; } c;
    c.v = __builtin_amdgcn_cvt_pk_bf16_f32(a, b);      // v_cvt_pk_bf16_f32
    return c.u;
#else
    return (unsigned)f2bf(a) | ((unsigned)f2bf(b) << 16);
#endif
}

// Async global->LDS 16-byte copy (ASYNCcnt tracked). LDS operand is the
// wave-relative byte offset = low 32 bits of the generic shared pointer.
__device__ __forceinline__ void async_copy16(const void* gptr, void* lptr) {
    unsigned ldsoff = (unsigned)(size_t)lptr;
    unsigned long long ga = (unsigned long long)(size_t)gptr;
    asm volatile("global_load_async_to_lds_b128 %0, %1, off"
                 :: "v"(ldsoff), "v"(ga) : "memory");
}
__device__ __forceinline__ void async_wait0() {
    asm volatile("s_wait_asynccnt 0x0" ::: "memory");
}

// A-matrix fragment, 16x32 bf16 (ISA 7.12.2): lane L holds row M=L&15,
// VGPR j half p -> K = ((j>=4)?16:0) + (j&3)*2 + (L>>4)*8 + p  (pairs contiguous)
__device__ __forceinline__ v16bf frag_a(const unsigned short* base, int ld, int lane) {
    FragBF r;
    const unsigned short* p = base + (lane & 15) * ld;
    const int g8 = (lane >> 4) << 3;
#pragma unroll
    for (int j = 0; j < 8; ++j) {
        int k = ((j & 4) << 2) + ((j & 3) << 1) + g8;
        r.u[j] = *(const unsigned*)(p + k);
    }
    return r.v;
}

// B-matrix fragment, 32x16 bf16: lane L holds column N=L&15,
// K = ((j>=4)?16:0) + (j&3)*2 + (L>>4)*8 + p.  LDS tile stored as Bt[N][K] so pairs contiguous.
__device__ __forceinline__ v16bf frag_b(const unsigned short* base, int ld, int lane) {
    FragBF r;
    const unsigned short* p = base + (lane & 15) * ld + ((lane >> 4) << 3);
#pragma unroll
    for (int j = 0; j < 8; ++j) r.u[j] = *(const unsigned*)(p + ((j & 4) << 2) + ((j & 3) << 1));
    return r.v;
}

__device__ __forceinline__ v8f wmma_bf16(v16bf a, v16bf b, v8f c) {
    return __builtin_amdgcn_wmma_f32_16x16x32_bf16(false, a, false, b, (short)0, c,
                                                   false, false);
}

// ---------------------------------------------------------------------------
// Kernel 0: streaming f32 -> bf16 conversion (8 elements / thread).
// ---------------------------------------------------------------------------
__global__ __launch_bounds__(256) void cvt_bf16_kernel(
    const float* __restrict__ src, unsigned short* __restrict__ dst, int n8)
{
    int i = blockIdx.x * 256 + threadIdx.x;
    if (i >= n8) return;
    size_t off = (size_t)i * 8;
    float4 a = *(const float4*)(src + off);
    float4 b = *(const float4*)(src + off + 4);
    uint4 o;
    o.x = pack_bf16(a.x, a.y);
    o.y = pack_bf16(a.z, a.w);
    o.z = pack_bf16(b.x, b.y);
    o.w = pack_bf16(b.z, b.w);
    *(uint4*)(dst + off) = o;
}

// ---------------------------------------------------------------------------
// Kernel 1: fused QKV projection.  C[m,n] = X[m,:] . Wcat[n,:]
// M=8192, K=1024, N=3072 (Wcat = [Wq;Wk;Wv] rows, bf16).
// Pure async-fed WMMA loop; output bf16 scattered into [B, NH, S, D].
// ---------------------------------------------------------------------------
__global__ __launch_bounds__(256) void qkv_gemm_kernel(
    const unsigned short* __restrict__ xb,
    const unsigned short* __restrict__ wc,
    unsigned short* __restrict__ qo, unsigned short* __restrict__ ko,
    unsigned short* __restrict__ vo)
{
    __shared__ unsigned short As[64][40];   // 64 m-rows x 32 k (+8 pad)
    __shared__ unsigned short Bs[64][40];   // 64 n-rows x 32 k (Bt layout)

    const int tid  = threadIdx.x;
    const int lane = tid & 31, wave = tid >> 5;
    const int m0 = blockIdx.y * 64;
    const int n0 = blockIdx.x * 64;
    const int wsel = n0 >> 10;
    unsigned short* outp = (wsel == 0) ? qo : (wsel == 1) ? ko : vo;

    const int tr = wave >> 2, tc = wave & 3;   // wave -> 2 row-tiles x 1 col-tile
    v8f acc0 = {}, acc1 = {};

    const int row = tid >> 2, c8 = (tid & 3) * 8;   // one b128 per thread per tile
    for (int kb = 0; kb < 1024; kb += 32) {
        async_copy16(xb + (size_t)(m0 + row) * 1024 + kb + c8, &As[row][c8]);
        async_copy16(wc + (size_t)(n0 + row) * 1024 + kb + c8, &Bs[row][c8]);
        if (kb + 32 < 1024) {   // global_prefetch_b8 hints for next K-block
            __builtin_prefetch(xb + (size_t)(m0 + row) * 1024 + kb + 32 + c8, 0, 3);
            __builtin_prefetch(wc + (size_t)(n0 + row) * 1024 + kb + 32 + c8, 0, 3);
        }
        async_wait0();
        __syncthreads();
        v16bf fa0 = frag_a(&As[(2 * tr + 0) * 16][0], 40, lane);
        v16bf fa1 = frag_a(&As[(2 * tr + 1) * 16][0], 40, lane);
        v16bf fb  = frag_b(&Bs[tc * 16][0], 40, lane);
        acc0 = wmma_bf16(fa0, fb, acc0);
        acc1 = wmma_bf16(fa1, fb, acc1);
        __syncthreads();
    }

    // epilogue: scatter bf16 into [B, NH, S, D]
    const int colg = n0 + tc * 16 + (lane & 15);
    const int head = (colg & 1023) >> 6;
    const int d    = colg & 63;
#pragma unroll
    for (int tt = 0; tt < 2; ++tt) {
        const v8f& a = tt ? acc1 : acc0;
#pragma unroll
        for (int j = 0; j < 8; ++j) {
            int m = m0 + (2 * tr + tt) * 16 + j + ((lane >> 4) << 3);
            int b = m >> 11, s = m & 2047;
            outp[((((size_t)b * 16 + head) * 2048 + s) << 6) + d] = f2bf(a[j]);
        }
    }
}

// ---------------------------------------------------------------------------
// Kernel 2: flash attention with ALiBi (slope = -1) + causal mask.
// One block per (b, head, 64 query rows). 8 waves; wave w owns query
// row-group r = w>>1 and two 16-wide d column tiles.
// ---------------------------------------------------------------------------
__global__ __launch_bounds__(256) void attn_kernel(
    const unsigned short* __restrict__ q,
    const unsigned short* __restrict__ k,
    const unsigned short* __restrict__ v,
    unsigned short* __restrict__ hws)
{
    __shared__ unsigned short Qs[64][72];   // 64 q-rows x 64 d (+8)
    __shared__ unsigned short Ks[32][72];   // 32 keys  x 64 d (Bt for scores)
    __shared__ unsigned short Vs[32][72];   // V staging (row-major, async target)
    __shared__ unsigned short Vt[64][40];   // 64 d x 32 keys (Bt for PV)
    __shared__ float          Ss[64][36];   // f32 scores
    __shared__ unsigned short Ps[64][40];   // bf16 probabilities (A for PV)
    __shared__ float mrow[64], lrow[64], arow[64];

    const int tid = threadIdx.x, lane = tid & 31, wave = tid >> 5;
    const int qs0  = blockIdx.x * 64;
    const int head = blockIdx.y, b = blockIdx.z;
    const size_t base = ((size_t)(b * 16 + head)) * 2048 * 64;
    const unsigned short* qb = q + base;
    const unsigned short* kbp = k + base;
    const unsigned short* vb = v + base;

    // async-load Q tile once (already bf16)
#pragma unroll
    for (int t = 0; t < 2; ++t) {
        int idx = tid + t * 256;
        int row = idx >> 3, c8 = (idx & 7) * 8;
        async_copy16(qb + (size_t)(qs0 + row) * 64 + c8, &Qs[row][c8]);
    }
    if (tid < 64) { mrow[tid] = -__builtin_inff(); lrow[tid] = 0.0f; }
    async_wait0();

    const int r  = wave >> 1;          // query row-group 0..3
    const int kc = wave & 1;           // score key-tile   0..1
    const int c0 = (wave & 1) * 2;     // PV d-tile pair
    v8f acc0 = {}, acc1 = {};

    const int kend = qs0 + 64;         // causal bound (multiple of 32)
    for (int kb0 = 0; kb0 < kend; kb0 += 32) {
        __syncthreads();               // protect prior-iteration LDS readers
        {   // async K / V tiles, 32x64 bf16 each (one b128 per thread per tile)
            int row = tid >> 3, c8 = (tid & 7) * 8;
            async_copy16(kbp + (size_t)(kb0 + row) * 64 + c8, &Ks[row][c8]);
            async_copy16(vb  + (size_t)(kb0 + row) * 64 + c8, &Vs[row][c8]);
            if (kb0 + 32 < kend) {
                __builtin_prefetch(kbp + (size_t)(kb0 + 32 + row) * 64 + c8, 0, 3);
                __builtin_prefetch(vb  + (size_t)(kb0 + 32 + row) * 64 + c8, 0, 3);
            }
        }
        async_wait0();
        __syncthreads();

        // ---- scores: S = Q K^T / sqrt(D) + (i - j), causal mask ----
        {
            v8f sc = {};
#pragma unroll
            for (int kd = 0; kd < 64; kd += 32) {
                v16bf fa = frag_a(&Qs[r * 16][kd], 72, lane);
                v16bf fb = frag_b(&Ks[kc * 16][kd], 72, lane);
                sc = wmma_bf16(fa, fb, sc);
            }
            const int tpos = kb0 + kc * 16 + (lane & 15);
#pragma unroll
            for (int j = 0; j < 8; ++j) {
                int rr = r * 16 + j + ((lane >> 4) << 3);
                int ipos = qs0 + rr;
                float val = sc[j] * 0.125f + (float)(ipos - tpos);
                if (tpos > ipos) val = -__builtin_inff();
                Ss[rr][kc * 16 + (lane & 15)] = val;
            }
        }
        __syncthreads();

        // ---- online softmax (tid<64) overlapped with V transpose (tid>=64) ----
        if (tid < 64) {
            float mo = mrow[tid];
            float mx = mo;
            float s[32];
#pragma unroll
            for (int c = 0; c < 32; ++c) { s[c] = Ss[tid][c]; mx = fmaxf(mx, s[c]); }
            float alpha = __expf(mo - mx);
            float sum = 0.0f;
#pragma unroll
            for (int c = 0; c < 32; ++c) {
                float p = __expf(s[c] - mx);
                sum += p;
                Ps[tid][c] = f2bf(p);
            }
            mrow[tid] = mx;
            lrow[tid] = lrow[tid] * alpha + sum;
            arow[tid] = alpha;
        } else {
            // transpose Vs[32][64] -> Vt[64][32] with the otherwise-idle threads
            for (int idx = tid - 64; idx < 512; idx += 192) {
                int kt = idx >> 4, c4 = (idx & 15) * 4;
                unsigned short e0 = Vs[kt][c4 + 0];
                unsigned short e1 = Vs[kt][c4 + 1];
                unsigned short e2 = Vs[kt][c4 + 2];
                unsigned short e3 = Vs[kt][c4 + 3];
                Vt[c4 + 0][kt] = e0; Vt[c4 + 1][kt] = e1;
                Vt[c4 + 2][kt] = e2; Vt[c4 + 3][kt] = e3;
            }
        }
        __syncthreads();

        // ---- PV: acc = acc*alpha + P @ V ----
        {
#pragma unroll
            for (int j = 0; j < 8; ++j) {
                float a = arow[r * 16 + j + ((lane >> 4) << 3)];
                acc0[j] *= a; acc1[j] *= a;
            }
            v16bf fa  = frag_a(&Ps[r * 16][0], 40, lane);
            v16bf fb0 = frag_b(&Vt[(c0 + 0) * 16][0], 40, lane);
            v16bf fb1 = frag_b(&Vt[(c0 + 1) * 16][0], 40, lane);
            acc0 = wmma_bf16(fa, fb0, acc0);
            acc1 = wmma_bf16(fa, fb1, acc1);
        }
    }
    __syncthreads();

    // epilogue: heads[b][s][head][d] in bf16
    const int d0 = (c0 + 0) * 16 + (lane & 15);
    const int d1 = (c0 + 1) * 16 + (lane & 15);
#pragma unroll
    for (int j = 0; j < 8; ++j) {
        int rr = r * 16 + j + ((lane >> 4) << 3);
        float inv = 1.0f / lrow[rr];
        size_t ob = (((size_t)b * 2048 + (qs0 + rr)) * 16 + head) * 64;
        hws[ob + d0] = f2bf(acc0[j] * inv);
        hws[ob + d1] = f2bf(acc1[j] * inv);
    }
}

// ---------------------------------------------------------------------------
// Kernel 3: output projection  out = heads @ Wo^T + bo  (f32 output).
// M=8192, K=1024, N=1024.  Both tiles bf16 -> pure async-fed WMMA loop.
// ---------------------------------------------------------------------------
__global__ __launch_bounds__(256) void out_gemm_kernel(
    const unsigned short* __restrict__ h,
    const unsigned short* __restrict__ wob,
    const float* __restrict__ bo,
    float* __restrict__ out)
{
    __shared__ unsigned short As[64][40];
    __shared__ unsigned short Bs[64][40];

    const int tid = threadIdx.x, lane = tid & 31, wave = tid >> 5;
    const int m0 = blockIdx.y * 64, n0 = blockIdx.x * 64;
    const int tr = wave >> 2, tc = wave & 3;
    v8f acc0 = {}, acc1 = {};

    const int row = tid >> 2, c8 = (tid & 3) * 8;
    for (int kb = 0; kb < 1024; kb += 32) {
        async_copy16(h   + (size_t)(m0 + row) * 1024 + kb + c8, &As[row][c8]);
        async_copy16(wob + (size_t)(n0 + row) * 1024 + kb + c8, &Bs[row][c8]);
        if (kb + 32 < 1024) {
            __builtin_prefetch(h   + (size_t)(m0 + row) * 1024 + kb + 32 + c8, 0, 3);
            __builtin_prefetch(wob + (size_t)(n0 + row) * 1024 + kb + 32 + c8, 0, 3);
        }
        async_wait0();
        __syncthreads();
        v16bf fa0 = frag_a(&As[(2 * tr + 0) * 16][0], 40, lane);
        v16bf fa1 = frag_a(&As[(2 * tr + 1) * 16][0], 40, lane);
        v16bf fb  = frag_b(&Bs[tc * 16][0], 40, lane);
        acc0 = wmma_bf16(fa0, fb, acc0);
        acc1 = wmma_bf16(fa1, fb, acc1);
        __syncthreads();
    }

    const int col = n0 + tc * 16 + (lane & 15);
    const float bias = bo[col];
#pragma unroll
    for (int tt = 0; tt < 2; ++tt) {
        const v8f& a = tt ? acc1 : acc0;
#pragma unroll
        for (int j = 0; j < 8; ++j) {
            int orow = m0 + (2 * tr + tt) * 16 + j + ((lane >> 4) << 3);
            out[(size_t)orow * 1024 + col] = a[j] + bias;
        }
    }
}

// ---------------------------------------------------------------------------
// Launch
// ---------------------------------------------------------------------------
extern "C" void kernel_launch(void* const* d_in, const int* in_sizes, int n_in,
                              void* d_out, int out_size, void* d_ws, size_t ws_size,
                              hipStream_t stream) {
    (void)in_sizes; (void)n_in; (void)out_size; (void)ws_size;
    const float* x  = (const float*)d_in[0];
    // d_in[1] is the causal mask (int32); causality is computed analytically.
    const float* Wq = (const float*)d_in[2];
    const float* Wk = (const float*)d_in[3];
    const float* Wv = (const float*)d_in[4];
    const float* Wo = (const float*)d_in[5];
    const float* bo = (const float*)d_in[6];
    float* out = (float*)d_out;

    const size_t QN = (size_t)4 * 16 * 2048 * 64;   // 8,388,608 elems per tensor
    const size_t WN = (size_t)1024 * 1024;          // 1,048,576 elems per weight
    unsigned short* qws  = (unsigned short*)d_ws;
    unsigned short* kws  = qws + QN;
    unsigned short* vws  = kws + QN;
    unsigned short* hws  = vws + QN;                // heads [B, S, NH, D]
    unsigned short* xb   = hws + QN;                // x in bf16 [8192,1024]
    unsigned short* wcat = xb + QN;                 // [Wq;Wk;Wv] bf16 [3072,1024]
    unsigned short* wob  = wcat + 3 * WN;           // Wo bf16 [1024,1024]

    // (0) streaming conversions
    cvt_bf16_kernel<<<dim3(4096), 256, 0, stream>>>(x,  xb,            (int)(QN / 8));
    cvt_bf16_kernel<<<dim3(512),  256, 0, stream>>>(Wq, wcat,          (int)(WN / 8));
    cvt_bf16_kernel<<<dim3(512),  256, 0, stream>>>(Wk, wcat + WN,     (int)(WN / 8));
    cvt_bf16_kernel<<<dim3(512),  256, 0, stream>>>(Wv, wcat + 2 * WN, (int)(WN / 8));
    cvt_bf16_kernel<<<dim3(512),  256, 0, stream>>>(Wo, wob,           (int)(WN / 8));

    // (1) QKV projections, (2) attention, (3) output projection
    qkv_gemm_kernel<<<dim3(48, 128), 256, 0, stream>>>(xb, wcat, qws, kws, vws);
    attn_kernel<<<dim3(32, 16, 4), 256, 0, stream>>>(qws, kws, vws, hws);
    out_gemm_kernel<<<dim3(16, 128), 256, 0, stream>>>(hws, wob, bo, out);
}
